// BiasedMultiHeadAtten_953482739905
// MI455X (gfx1250) — compile-verified, matrix operand order checked by reading, not run on hardware
//
#include <hip/hip_runtime.h>

typedef __bf16 bf16_t;
typedef __attribute__((ext_vector_type(16))) __bf16 bf16x16;
typedef __attribute__((ext_vector_type(8)))  __bf16 bf16x8;
typedef __attribute__((ext_vector_type(4)))  __bf16 bf16x4;
typedef __attribute__((ext_vector_type(8)))  float  f32x8;

__device__ __forceinline__ f32x8 zero8() {
    f32x8 z;
    #pragma unroll
    for (int i = 0; i < 8; ++i) z[i] = 0.0f;
    return z;
}

// Permutation of a 32-wide K-group so each WMMA bf16 fragment is one
// contiguous 32-byte LDS read per lane:
//   lanes 0-15 need K {0..7,16..23}, lanes 16-31 need K {8..15,24..31}.
__device__ __forceinline__ int permk(int k) {
    return (k & 7) + ((k >> 3) & 1) * 16 + ((k >> 4) & 1) * 8;
}

// ---------------------------------------------------------------------------
// GEMM core: C[M,N] = A[M,K] @ W[N,K]^T  (A,W f32 in global, bf16 in LDS)
// Block tile 128(M) x 64(N), K-step 32, 8 waves each computing 32x32.
// Software-pipelined: tile k+1 global loads overlap tile k WMMAs.
// ---------------------------------------------------------------------------
#define BM 128
#define BN 64
#define BK 32

// epilogue modes for gemm_wmma_epi
#define EPI_LIN   0   // out = acc + bias
#define EPI_GATE  1   // out = aux * sigmoid(acc + bias)
#define EPI_FINAL 2   // out = acc + bias + aux

// Shared GEMM mainloop body as a macro so both kernels keep identical codegen.
// Declares/updates: c00..c11 accumulators, As/Bs LDS tiles.
#define GEMM_MAINLOOP(A_, W_, K_)                                              \
    float4 ra[4], rb[2];                                                       \
    /* prologue: load tile 0 */                                                \
    _Pragma("unroll")                                                          \
    for (int i = 0; i < 4; ++i) {                                              \
        int c = tid + i * 256; int row = c >> 3; int kk = (c & 7) << 2;        \
        ra[i] = *(const float4*)(A_ + (size_t)(bm + row) * K_ + kk);           \
    }                                                                          \
    _Pragma("unroll")                                                          \
    for (int i = 0; i < 2; ++i) {                                              \
        int c = tid + i * 256; int row = c >> 3; int kk = (c & 7) << 2;        \
        rb[i] = *(const float4*)(W_ + (size_t)(bn + row) * K_ + kk);           \
    }                                                                          \
    for (int k0 = 0; k0 < K_; k0 += BK) {                                      \
        __syncthreads();                                                       \
        /* convert + store current tile to LDS */                              \
        _Pragma("unroll")                                                      \
        for (int i = 0; i < 4; ++i) {                                          \
            int c = tid + i * 256; int row = c >> 3; int kk = (c & 7) << 2;    \
            bf16x4 b; b[0] = (bf16_t)ra[i].x; b[1] = (bf16_t)ra[i].y;          \
            b[2] = (bf16_t)ra[i].z; b[3] = (bf16_t)ra[i].w;                    \
            *(bf16x4*)&As[row][permk(kk)] = b;                                 \
        }                                                                      \
        _Pragma("unroll")                                                      \
        for (int i = 0; i < 2; ++i) {                                          \
            int c = tid + i * 256; int row = c >> 3; int kk = (c & 7) << 2;    \
            bf16x4 b; b[0] = (bf16_t)rb[i].x; b[1] = (bf16_t)rb[i].y;          \
            b[2] = (bf16_t)rb[i].z; b[3] = (bf16_t)rb[i].w;                    \
            *(bf16x4*)&Bs[row][permk(kk)] = b;                                 \
        }                                                                      \
        __syncthreads();                                                       \
        /* issue next tile's loads: overlap with fragment loads + WMMAs */     \
        int knext = k0 + BK;                                                   \
        if (knext < K_) {                                                      \
            _Pragma("unroll")                                                  \
            for (int i = 0; i < 4; ++i) {                                      \
                int c = tid + i * 256; int row = c >> 3; int kk = (c & 7) << 2;\
                ra[i] = *(const float4*)(A_ + (size_t)(bm + row) * K_ + knext + kk); \
            }                                                                  \
            _Pragma("unroll")                                                  \
            for (int i = 0; i < 2; ++i) {                                      \
                int c = tid + i * 256; int row = c >> 3; int kk = (c & 7) << 2;\
                rb[i] = *(const float4*)(W_ + (size_t)(bn + row) * K_ + knext + kk); \
            }                                                                  \
        }                                                                      \
        /* prefetch tile after next toward L2/WGP$ (global_prefetch_b8) */     \
        int kpf = k0 + 2 * BK;                                                 \
        if (kpf < K_) {                                                        \
            if (tid < 128)      __builtin_prefetch(A_ + (size_t)(bm + tid) * K_ + kpf, 0, 3); \
            else if (tid < 192) __builtin_prefetch(W_ + (size_t)(bn + tid - 128) * K_ + kpf, 0, 3); \
        }                                                                      \
        bf16x16 a0 = *(const bf16x16*)&As[wm + r][half];                       \
        bf16x16 a1 = *(const bf16x16*)&As[wm + 16 + r][half];                  \
        bf16x16 b0 = *(const bf16x16*)&Bs[wn + r][half];                       \
        bf16x16 b1 = *(const bf16x16*)&Bs[wn + 16 + r][half];                  \
        c00 = __builtin_amdgcn_wmma_f32_16x16x32_bf16(false, a0, false, b0, (short)0, c00, false, false); \
        c01 = __builtin_amdgcn_wmma_f32_16x16x32_bf16(false, a0, false, b1, (short)0, c01, false, false); \
        c10 = __builtin_amdgcn_wmma_f32_16x16x32_bf16(false, a1, false, b0, (short)0, c10, false, false); \
        c11 = __builtin_amdgcn_wmma_f32_16x16x32_bf16(false, a1, false, b1, (short)0, c11, false, false); \
    }

__global__ __launch_bounds__(256)
void gemm_wmma_epi(const float* __restrict__ A, const float* __restrict__ W,
                   const float* __restrict__ bias, const float* __restrict__ aux,
                   float* __restrict__ out, int M, int N, int K, int mode) {
    __shared__ __attribute__((aligned(32))) bf16_t As[BM][BK];
    __shared__ __attribute__((aligned(32))) bf16_t Bs[BN][BK];

    const int tid  = threadIdx.x;
    const int lane = tid & 31;
    const int w    = tid >> 5;
    const int wm   = (w & 3) * 32;   // 4 wave-rows of 32
    const int wn   = (w >> 2) * 32;  // 2 wave-cols of 32
    const int bm   = blockIdx.x * BM;
    const int bn   = blockIdx.y * BN;
    const int r    = lane & 15;
    const int half = (lane >> 4) << 4;

    f32x8 c00 = zero8(), c01 = zero8(), c10 = zero8(), c11 = zero8();

    GEMM_MAINLOOP(A, W, K)

    f32x8 cs[2][2]; cs[0][0] = c00; cs[0][1] = c01; cs[1][0] = c10; cs[1][1] = c11;
    const int rh = lane >> 4;
    #pragma unroll
    for (int fm = 0; fm < 2; ++fm)
        #pragma unroll
        for (int fn = 0; fn < 2; ++fn) {
            int col = bn + wn + fn * 16 + r;
            float bv = bias[col];
            #pragma unroll
            for (int t = 0; t < 8; ++t) {
                int row = bm + wm + fm * 16 + t + rh * 8;
                float v = cs[fm][fn][t] + bv;
                size_t idx = (size_t)row * N + col;
                if (mode == EPI_LIN) {
                    out[idx] = v;
                } else if (mode == EPI_GATE) {
                    float sg = __builtin_amdgcn_rcpf(1.0f + __expf(-v));  // v_rcp_f32
                    out[idx] = aux[idx] * sg;
                } else {
                    out[idx] = v + aux[idx];
                }
            }
        }
}

// ---------------------------------------------------------------------------
// Same GEMM core, epilogue writes bf16 into the scrambled head layout
//   q[l,h] -> scr[head=(l%1024)/64][pos=2h+l/1024][dim=l%64]
// ---------------------------------------------------------------------------
__global__ __launch_bounds__(256)
void gemm_wmma_qk(const float* __restrict__ A, const float* __restrict__ W,
                  const float* __restrict__ bias, bf16_t* __restrict__ out_scr,
                  int M, int N, int K) {
    __shared__ __attribute__((aligned(32))) bf16_t As[BM][BK];
    __shared__ __attribute__((aligned(32))) bf16_t Bs[BN][BK];

    const int tid  = threadIdx.x;
    const int lane = tid & 31;
    const int w    = tid >> 5;
    const int wm   = (w & 3) * 32;
    const int wn   = (w >> 2) * 32;
    const int bm   = blockIdx.x * BM;
    const int bn   = blockIdx.y * BN;
    const int r    = lane & 15;
    const int half = (lane >> 4) << 4;

    f32x8 c00 = zero8(), c01 = zero8(), c10 = zero8(), c11 = zero8();

    GEMM_MAINLOOP(A, W, K)

    f32x8 cs[2][2]; cs[0][0] = c00; cs[0][1] = c01; cs[1][0] = c10; cs[1][1] = c11;
    const int rh = lane >> 4;
    #pragma unroll
    for (int fm = 0; fm < 2; ++fm)
        #pragma unroll
        for (int fn = 0; fn < 2; ++fn) {
            int col = bn + wn + fn * 16 + r;          // h
            float bv = bias[col];
            #pragma unroll
            for (int t = 0; t < 8; ++t) {
                int row = bm + wm + fm * 16 + t + rh * 8;   // l
                float v = cs[fm][fn][t] + bv;
                int head = (row & 1023) >> 6;
                int pos  = 2 * col + (row >> 10);
                int dim  = row & 63;
                out_scr[((size_t)head << 17) + (size_t)pos * 64 + dim] = (bf16_t)v;
            }
        }
}

// ---------------------------------------------------------------------------
// Flash-attention: grid (16 qblocks, 16 heads), 8 waves, 16 q-rows/wave.
// Scores kept in registers (268 MB score tensor > 192 MB L2 -> never stored).
// ---------------------------------------------------------------------------
__global__ __launch_bounds__(256)
void attn_fa(const bf16_t* __restrict__ q_scr, const bf16_t* __restrict__ k_scr,
             float* __restrict__ o) {
    const int L = 2048;
    const int head = blockIdx.y;
    const int tid = threadIdx.x, lane = tid & 31, w = tid >> 5;
    const int r = lane & 15, rh = lane >> 4;
    const int qbase = blockIdx.x * 128 + w * 16;

    const bf16_t* qh = q_scr + ((size_t)head << 17);
    const bf16_t* kh = k_scr + ((size_t)head << 17);   // V == K

    __shared__ __attribute__((aligned(32))) bf16_t Vt[64][32];       // [dim][permk(key)]
    __shared__ __attribute__((aligned(32))) bf16_t Plds[8][16][32];  // per-wave P bounce

    // Q A-fragments: 16 rows x 64 dims = 2 fragments of 16x32
    bf16x16 qa[2];
    {
        const bf16_t* qrow = qh + (size_t)(qbase + r) * 64;
        #pragma unroll
        for (int f = 0; f < 2; ++f) {
            bf16x8 lo = *(const bf16x8*)(qrow + f * 32 + rh * 8);
            bf16x8 hi = *(const bf16x8*)(qrow + f * 32 + rh * 8 + 16);
            qa[f] = __builtin_shufflevector(lo, hi, 0,1,2,3,4,5,6,7,8,9,10,11,12,13,14,15);
        }
    }

    float mrow[8], lrow[8];
    f32x8 acc[4];
    #pragma unroll
    for (int t = 0; t < 8; ++t) { mrow[t] = -1e30f; lrow[t] = 0.0f; }
    #pragma unroll
    for (int f = 0; f < 4; ++f) acc[f] = zero8();

    const float scale = 0.125f;   // 64^-0.5

    for (int j = 0; j < L; j += 32) {
        // cooperative stage of transposed V tile: (key,dim) -> Vt[dim][permk(key)]
        // batch the loads first so they issue back-to-back
        bf16x4 rv[2];
        #pragma unroll
        for (int i = 0; i < 2; ++i) {
            int c   = tid + i * 256;        // 512 chunks of 4 dims
            int key = c >> 4;
            int d4  = (c & 15) << 2;
            rv[i] = *(const bf16x4*)(kh + (size_t)(j + key) * 64 + d4);
        }
        __syncthreads();
        #pragma unroll
        for (int i = 0; i < 2; ++i) {
            int c   = tid + i * 256;
            int key = c >> 4;
            int d4  = (c & 15) << 2;
            int pk  = permk(key);
            Vt[d4 + 0][pk] = rv[i][0]; Vt[d4 + 1][pk] = rv[i][1];
            Vt[d4 + 2][pk] = rv[i][2]; Vt[d4 + 3][pk] = rv[i][3];
        }
        __syncthreads();

        // prefetch next key tile (32 keys x 64 dims = 4 KB)
        if (j + 32 < L && tid < 32)
            __builtin_prefetch(kh + (size_t)(j + 32 + tid) * 64, 0, 3);

        // S = (Q*scale) @ K^T for 32 keys: two 16x16 C tiles
        f32x8 s[2]; s[0] = zero8(); s[1] = zero8();
        #pragma unroll
        for (int f = 0; f < 2; ++f) {          // dim half
            #pragma unroll
            for (int n = 0; n < 2; ++n) {      // key subtile
                const bf16_t* krow = kh + (size_t)(j + n * 16 + r) * 64;
                bf16x8 lo = *(const bf16x8*)(krow + f * 32 + rh * 8);
                bf16x8 hi = *(const bf16x8*)(krow + f * 32 + rh * 8 + 16);
                bf16x16 kb = __builtin_shufflevector(lo, hi, 0,1,2,3,4,5,6,7,8,9,10,11,12,13,14,15);
                s[n] = __builtin_amdgcn_wmma_f32_16x16x32_bf16(false, qa[f], false, kb,
                                                               (short)0, s[n], false, false);
            }
        }
        #pragma unroll
        for (int n = 0; n < 2; ++n)
            #pragma unroll
            for (int t = 0; t < 8; ++t) s[n][t] *= scale;

        // online softmax (rows live across 16-lane halves)
        float pm[8];
        #pragma unroll
        for (int t = 0; t < 8; ++t) pm[t] = fmaxf(s[0][t], s[1][t]);
        #pragma unroll
        for (int off = 8; off > 0; off >>= 1)
            #pragma unroll
            for (int t = 0; t < 8; ++t) pm[t] = fmaxf(pm[t], __shfl_xor(pm[t], off, 32));

        float alpha[8], ps[8];
        f32x8 p0, p1;
        #pragma unroll
        for (int t = 0; t < 8; ++t) {
            float mn = fmaxf(mrow[t], pm[t]);
            alpha[t] = __expf(mrow[t] - mn);
            mrow[t] = mn;
            p0[t] = __expf(s[0][t] - mn);
            p1[t] = __expf(s[1][t] - mn);
            ps[t] = p0[t] + p1[t];
        }
        #pragma unroll
        for (int off = 8; off > 0; off >>= 1)
            #pragma unroll
            for (int t = 0; t < 8; ++t) ps[t] += __shfl_xor(ps[t], off, 32);
        #pragma unroll
        for (int t = 0; t < 8; ++t) lrow[t] = lrow[t] * alpha[t] + ps[t];
        #pragma unroll
        for (int f = 0; f < 4; ++f)
            #pragma unroll
            for (int t = 0; t < 8; ++t) acc[f][t] *= alpha[t];

        // bounce P (C layout) -> A-fragment layout via per-wave LDS
        bf16_t* pl = &Plds[w][0][0];
        #pragma unroll
        for (int t = 0; t < 8; ++t) {
            int m = t + 8 * rh;
            pl[m * 32 + permk(r)]      = (bf16_t)p0[t];
            pl[m * 32 + permk(16 + r)] = (bf16_t)p1[t];
        }
        __builtin_amdgcn_wave_barrier();
        bf16x16 pa = *(const bf16x16*)&Plds[w][r][rh * 16];

        // O += P @ V   (4 dim subtiles)
        #pragma unroll
        for (int f = 0; f < 4; ++f) {
            bf16x16 vb = *(const bf16x16*)&Vt[f * 16 + r][rh * 16];
            acc[f] = __builtin_amdgcn_wmma_f32_16x16x32_bf16(false, pa, false, vb,
                                                             (short)0, acc[f], false, false);
        }
    }

    float inv[8];
    #pragma unroll
    for (int t = 0; t < 8; ++t) inv[t] = __builtin_amdgcn_rcpf(lrow[t]);
    #pragma unroll
    for (int f = 0; f < 4; ++f)
        #pragma unroll
        for (int t = 0; t < 8; ++t) {
            int row = qbase + t + 8 * rh;
            int col = head * 64 + f * 16 + r;
            o[(size_t)row * 1024 + col] = acc[f][t] * inv[t];
        }
}

extern "C" void kernel_launch(void* const* d_in, const int* in_sizes, int n_in,
                              void* d_out, int out_size, void* d_ws, size_t ws_size,
                              hipStream_t stream) {
    const float* NE   = (const float*)d_in[0];
    const float* AB   = (const float*)d_in[1];
    const float* Wq   = (const float*)d_in[2];
    const float* bq   = (const float*)d_in[3];
    const float* Wk   = (const float*)d_in[4];
    const float* bk   = (const float*)d_in[5];
    const float* Wlin = (const float*)d_in[6];
    const float* blin = (const float*)d_in[7];
    const float* Wres = (const float*)d_in[8];
    const float* bres = (const float*)d_in[9];
    const float* Wo   = (const float*)d_in[10];
    const float* bo   = (const float*)d_in[11];
    float* out = (float*)d_out;

    char* ws = (char*)d_ws;
    bf16_t* q_scr = (bf16_t*)(ws);                       //  4 MB  [16][2048][64] bf16
    bf16_t* k_scr = (bf16_t*)(ws + ((size_t)4  << 20));  //  4 MB
    float*  lin   = (float*) (ws + ((size_t)8  << 20));  //  8 MB  2048x1024 f32
    float*  res   = (float*) (ws + ((size_t)16 << 20));  //  8 MB
    float*  obuf  = (float*) (ws + ((size_t)24 << 20));  //  8 MB

    dim3 blk(256);
    dim3 g(2048 / BM, 1024 / BN);   // (16,16)

    // Q/K projections (write scrambled bf16 head layout)
    gemm_wmma_qk<<<g, blk, 0, stream>>>(NE, Wq, bq, q_scr, 2048, 1024, 4096);
    gemm_wmma_qk<<<g, blk, 0, stream>>>(NE, Wk, bk, k_scr, 2048, 1024, 4096);

    // attention (V = K)
    attn_fa<<<dim3(16, 16), blk, 0, stream>>>(q_scr, k_scr, obuf);

    // gated residual branch
    gemm_wmma_epi<<<g, blk, 0, stream>>>(AB,  Wlin, blin, nullptr, lin, 2048, 1024, 2048, EPI_LIN);
    gemm_wmma_epi<<<g, blk, 0, stream>>>(lin, Wres, bres, lin,     res, 2048, 1024, 1024, EPI_GATE);

    // output projection + bias + residual
    gemm_wmma_epi<<<g, blk, 0, stream>>>(obuf, Wo, bo, res, out, 2048, 1024, 1024, EPI_FINAL);
}
